// EdgeHead_66975720014384
// MI455X (gfx1250) — compile-verified, matrix operand order checked by reading, not run on hardware
//
#include <hip/hip_runtime.h>

typedef __attribute__((ext_vector_type(16))) __bf16 v16bf;
typedef __attribute__((ext_vector_type(8)))  float  v8f;

#define TILE_E   128
#define NTHREADS 256
#define MAXBLOCKS 2048

// Padded row strides (bytes) to avoid LDS bank conflicts:
//  SBIG = 528  (132 dw, 132 % 64 = 4  -> lane r hits banks 4r..4r+3, conflict-free)
//  SSML = 144  (36 dw,  gcd(36,64)=4  -> 16 distinct 4-bank groups)
#define SBIG 528
#define SSML 144

// ---------------- LDS layout (bytes) ----------------
#define SA_OFF    0                         // 128 x SBIG = 67584 (gathered edge feats bf16)
#define SH2_OFF   0                         // 128 x SSML = 18432 (h2 bf16, reuses sA)
#define SH3_OFF   18432                     // 128 x 16 f32 = 8192 (reuses sA)
#define SH_OFF    67584                     // 128 x SBIG = 67584 (h1 bf16)
#define W1T_OFF   135168                    // 256 x SBIG = 135168
#define W2T_OFF   270336                    //  64 x SBIG = 33792
#define W3T_OFF   304128                    //  16 x SSML = 2304
#define B1_OFF    306432
#define B2_OFF    307456
#define B3_OFF    307712
#define W4_OFF    307776
#define B4_OFF    307840
#define LDS_BYTES 307968

// workspace (global, bf16, pre-padded identically to LDS weight region)
#define W2T_WS    135168
#define W3T_WS    168960
#define WS_DWORDS 42816                     // 171264 bytes / 4

__device__ __forceinline__ float relu1(float x) {           // single v_med3
  return __builtin_amdgcn_fmed3f(x, 0.0f, __builtin_inff());
}

__device__ __forceinline__ unsigned pk2(float a, float b) {
  union { __bf16 h[2]; unsigned u; } t;
  t.h[0] = (__bf16)a; t.h[1] = (__bf16)b;
  return t.u;
}

// One WMMA A/B fragment: 16B at p and 16B at p+32
// (CDNA5 16-bit layout: VGPR0-3 hold K=kb..kb+7, VGPR4-7 hold K=kb+16..kb+23)
__device__ __forceinline__ v16bf ld_frag(const char* p) {
  union { uint4 u[2]; v16bf v; } t;
  t.u[0] = *(const uint4*)(p);
  t.u[1] = *(const uint4*)(p + 32);
  return t.v;
}

// -------- prep: W1,W2,W3 fp32 -> bf16 transposed+padded (Wt[n][k]) in d_ws --------
__global__ void edgehead_prep(const float* __restrict__ W1,
                              const float* __restrict__ W2,
                              const float* __restrict__ W3,
                              char* __restrict__ wsb) {
  int i = blockIdx.x * NTHREADS + threadIdx.x;
  if (i < 65536) {                       // W1^T: 256 x 256
    int n = i >> 8, k = i & 255;
    *(__bf16*)(wsb + n * SBIG + 2 * k) = (__bf16)W1[k * 256 + n];
  } else if (i < 65536 + 16384) {        // W2^T: 64 x 256
    int j = i - 65536; int n = j >> 8, k = j & 255;
    *(__bf16*)(wsb + W2T_WS + n * SBIG + 2 * k) = (__bf16)W2[k * 64 + n];
  } else if (i < 65536 + 16384 + 1024) { // W3^T: 16 x 64
    int j = i - 81920; int n = j >> 6, k = j & 63;
    *(__bf16*)(wsb + W3T_WS + n * SSML + 2 * k) = (__bf16)W3[k * 16 + n];
  }
}

// ------------------------------ main kernel ------------------------------
// Persistent blocks: weights staged into LDS ONCE, then grid-stride over edge tiles.
__global__ void edgehead_mlp(const float* __restrict__ z,
                             const int*   __restrict__ eidx,
                             const unsigned* __restrict__ ws,
                             const float* __restrict__ b1,
                             const float* __restrict__ b2,
                             const float* __restrict__ b3,
                             const float* __restrict__ W4,
                             const float* __restrict__ b4,
                             float* __restrict__ out,
                             int E, int ntiles) {
  extern __shared__ char smem[];
  const int tid  = threadIdx.x;
  const int wave = tid >> 5;
  const int lane = tid & 31;
  const int lane16 = lane & 15;
  const int kb2    = (lane >> 4) << 4;  // K-base bytes: lanes 16-31 start at K=8 (16B)
  const int rowadd = (lane >> 4) << 3;  // C/D rows: lanes 16-31 hold M = r + 8

  // ---- stage weights once: flat dword copy of padded bf16 W1^T|W2^T|W3^T ----
  {
    unsigned* dst = (unsigned*)(smem + W1T_OFF);
    for (int i = tid; i < WS_DWORDS; i += NTHREADS) dst[i] = ws[i];
    if (tid < 256) ((float*)(smem + B1_OFF))[tid] = b1[tid];
    if (tid < 64)  ((float*)(smem + B2_OFF))[tid] = b2[tid];
    if (tid < 16) {
      ((float*)(smem + B3_OFF))[tid] = b3[tid];
      ((float*)(smem + W4_OFF))[tid] = W4[tid];
    }
    if (tid == 0)  ((float*)(smem + B4_OFF))[0] = b4[0];
  }
  __syncthreads();

  for (int tile = blockIdx.x; tile < ntiles; tile += gridDim.x) {
    const int ebase = tile * TILE_E;

    // ---- gather: each thread fills one half-row (src or dst features) ----
    {
      int e_local = tid >> 1;
      int which   = tid & 1;               // 0 = src half, 1 = dst half
      int eg = ebase + e_local;
      if (eg >= E) eg = E - 1;             // clamp: keep EXEC all-ones for WMMA
      int node = eidx[which * E + eg];
      const float4* zr = (const float4*)(z + (size_t)node * 128);
      uint2* drow = (uint2*)(smem + SA_OFF + e_local * SBIG + which * 256);
#pragma unroll 8
      for (int i = 0; i < 32; ++i) {
        float4 f = zr[i];
        drow[i] = make_uint2(pk2(f.x, f.y), pk2(f.z, f.w));
      }
    }
    __syncthreads();

    // ---- GEMM1: h1 = relu(A[128x256] @ W1 + b1) -> sH bf16 ----
    {
      const char* sA  = smem + SA_OFF;
      const char* sW1 = smem + W1T_OFF;
      char*       sH  = smem + SH_OFF;
      const float* sB1 = (const float*)(smem + B1_OFF);
      for (int nt = wave * 2; nt < wave * 2 + 2; ++nt) {
        const int ncol = nt * 16 + lane16;
        const char* brow = sW1 + ncol * SBIG;
        v16bf bfr[8];
#pragma unroll
        for (int s = 0; s < 8; ++s) bfr[s] = ld_frag(brow + 64 * s + kb2);
        const float bias = sB1[ncol];
        for (int mt = 0; mt < 8; ++mt) {
          v8f c;
#pragma unroll
          for (int i = 0; i < 8; ++i) c[i] = bias;
          const char* arow = sA + (mt * 16 + lane16) * SBIG;
#pragma unroll
          for (int s = 0; s < 8; ++s) {
            v16bf a = ld_frag(arow + 64 * s + kb2);
            c = __builtin_amdgcn_wmma_f32_16x16x32_bf16(false, a, false, bfr[s],
                                                        (short)0, c, false, false);
          }
#pragma unroll
          for (int r = 0; r < 8; ++r) {
            int row = mt * 16 + r + rowadd;
            *(__bf16*)(sH + row * SBIG + ncol * 2) = (__bf16)relu1(c[r]);
          }
        }
      }
    }
    __syncthreads();

    // ---- GEMM2: h2 = relu(h1[128x256] @ W2 + b2) -> sH2 bf16 (reuses sA) ----
    {
      const char* sH  = smem + SH_OFF;
      const char* sW2 = smem + W2T_OFF;
      char*       sH2 = smem + SH2_OFF;
      const float* sB2 = (const float*)(smem + B2_OFF);
      const int nt  = wave >> 1;               // 4 n-tiles, 2 waves each
      const int mt0 = (wave & 1) * 4;          // 4 m-tiles per wave
      const int ncol = nt * 16 + lane16;
      const char* brow = sW2 + ncol * SBIG;
      v16bf bfr[8];
#pragma unroll
      for (int s = 0; s < 8; ++s) bfr[s] = ld_frag(brow + 64 * s + kb2);
      const float bias = sB2[ncol];
      for (int mt = mt0; mt < mt0 + 4; ++mt) {
        v8f c;
#pragma unroll
        for (int i = 0; i < 8; ++i) c[i] = bias;
        const char* arow = sH + (mt * 16 + lane16) * SBIG;
#pragma unroll
        for (int s = 0; s < 8; ++s) {
          v16bf a = ld_frag(arow + 64 * s + kb2);
          c = __builtin_amdgcn_wmma_f32_16x16x32_bf16(false, a, false, bfr[s],
                                                      (short)0, c, false, false);
        }
#pragma unroll
        for (int r = 0; r < 8; ++r) {
          int row = mt * 16 + r + rowadd;
          *(__bf16*)(sH2 + row * SSML + ncol * 2) = (__bf16)relu1(c[r]);
        }
      }
    }
    __syncthreads();

    // ---- GEMM3: h3 = relu(h2[128x64] @ W3 + b3) -> sH3 f32 ----
    {
      const char* sH2 = smem + SH2_OFF;
      const char* sW3 = smem + W3T_OFF;
      float*      sH3 = (float*)(smem + SH3_OFF);
      const float* sB3 = (const float*)(smem + B3_OFF);
      const int mt = wave;                      // 8 m-tiles, 1 per wave
      const int ncol = lane16;                  // single n-tile (N=16)
      const char* brow = sW3 + ncol * SSML;
      const char* arow = sH2 + (mt * 16 + lane16) * SSML;
      v8f c;
      const float bias = sB3[ncol];
#pragma unroll
      for (int i = 0; i < 8; ++i) c[i] = bias;
#pragma unroll
      for (int s = 0; s < 2; ++s) {
        v16bf a = ld_frag(arow + 64 * s + kb2);
        v16bf b = ld_frag(brow + 64 * s + kb2);
        c = __builtin_amdgcn_wmma_f32_16x16x32_bf16(false, a, false, b,
                                                    (short)0, c, false, false);
      }
#pragma unroll
      for (int r = 0; r < 8; ++r) {
        int row = mt * 16 + r + rowadd;
        sH3[row * 16 + ncol] = relu1(c[r]);     // relu before final linear
      }
    }
    __syncthreads();

    // ---- layer 4: sigmoid(h3 . W4 + b4), one thread per edge ----
    if (tid < TILE_E) {
      const float* h3 = (const float*)(smem + SH3_OFF) + tid * 16;
      const float* w4 = (const float*)(smem + W4_OFF);
      float s = ((const float*)(smem + B4_OFF))[0];
#pragma unroll
      for (int j = 0; j < 16; ++j) s += h3[j] * w4[j];
      float sig = 1.0f / (1.0f + __expf(-s));
      int eg = ebase + tid;
      if (eg < E) out[eg] = sig;
    }
    __syncthreads();  // next gather overwrites sA (aliases sH2/sH3)
  }
}

extern "C" void kernel_launch(void* const* d_in, const int* in_sizes, int n_in,
                              void* d_out, int out_size, void* d_ws, size_t ws_size,
                              hipStream_t stream) {
  const float* z    = (const float*)d_in[0];
  const int*   eidx = (const int*)d_in[1];   // edge_index [2, E]
  const float* W1   = (const float*)d_in[2];
  const float* b1   = (const float*)d_in[3];
  const float* W2   = (const float*)d_in[4];
  const float* b2   = (const float*)d_in[5];
  const float* W3   = (const float*)d_in[6];
  const float* b3   = (const float*)d_in[7];
  const float* W4   = (const float*)d_in[8];
  const float* b4   = (const float*)d_in[9];
  float* out = (float*)d_out;

  const int E = in_sizes[1] / 2;

  edgehead_prep<<<(82944 + NTHREADS - 1) / NTHREADS, NTHREADS, 0, stream>>>(
      W1, W2, W3, (char*)d_ws);

  const int ntiles  = (E + TILE_E - 1) / TILE_E;
  const int nblocks = ntiles < MAXBLOCKS ? ntiles : MAXBLOCKS;
  edgehead_mlp<<<nblocks, NTHREADS, LDS_BYTES, stream>>>(
      z, eidx, (const unsigned*)d_ws, b1, b2, b3, W4, b4, out, E, ntiles);
}
